// SelfAttention_4767413699263
// MI455X (gfx1250) — matrix operand order repaired and checked by reading, not compiled, not run on hardware
//
#include <hip/hip_runtime.h>
#include <hip/hip_bf16.h>

// ---------------------------------------------------------------------------
// Self-attention forward for MI455X (gfx1250), wave32 + WMMA f16.
//   B=4, S=2048, EMB=1024, H=16, D=64.
// All matmuls via v_wmma_f32_16x16x32_f16 (f32 accumulate).
// B operands stored transposed in LDS so every fragment load is contiguous
// (lowers to ds_load_b128 pairs instead of strided ds_load_u16 + packing).
// ---------------------------------------------------------------------------

#define BATCH 4
#define SEQ   2048
#define EMB   1024
#define HEADS 16
#define DH    64

typedef __attribute__((ext_vector_type(16))) _Float16 v16h;
typedef __attribute__((ext_vector_type(8)))  float    v8f;

static __device__ __forceinline__ v8f wmma16(v16h a, v16h b, v8f c) {
    // D = A(16x32 f16) * B(32x16 f16) + C(16x16 f32)
    return __builtin_amdgcn_wmma_f32_16x16x32_f16(
        false, a, false, b, (short)0, c, false, false);
}

// A-fragment (16x32, row-major source, leading dim ld in halfs).
// ISA: lanes 0-15 -> M=lane, elems 0..7 = K 0..7, elems 8..15 = K 16..23;
//      lanes 16-31 -> M=lane-16, K 8..15 / 24..31.
static __device__ __forceinline__ v16h frag_a(const _Float16* p, int ld) {
    int lane = threadIdx.x & 31;
    int m  = lane & 15;
    int kb = (lane >> 4) << 3;
    const _Float16* q = p + m * ld + kb;
    v16h f;
#pragma unroll
    for (int j = 0; j < 8; ++j) { f[j] = q[j]; f[j + 8] = q[16 + j]; }
    return f;
}

// B-fragment where the source T is row-major [N][K] (i.e. B = T^T, B is KxN).
// ISA B layout: lanes 0-15 hold K=0..7 / 16..23, lanes 16-31 K=8..15 / 24..31,
// N = lane&15.  Per-lane reads are two 8-half chunks (2x ds_load_b128).
static __device__ __forceinline__ v16h frag_bt(const _Float16* t, int ld) {
    int lane = threadIdx.x & 31;
    int n  = lane & 15;
    int kb = (lane >> 4) << 3;
    const _Float16* q = t + n * ld + kb;
    v16h f;
#pragma unroll
    for (int j = 0; j < 8; ++j) { f[j] = q[j]; f[j + 8] = q[16 + j]; }
    return f;
}

static __device__ __forceinline__ float redmax16(float x) {
    x = fmaxf(x, __shfl_xor(x, 1, 32));
    x = fmaxf(x, __shfl_xor(x, 2, 32));
    x = fmaxf(x, __shfl_xor(x, 4, 32));
    x = fmaxf(x, __shfl_xor(x, 8, 32));
    return x;
}
static __device__ __forceinline__ float redsum16(float x) {
    x += __shfl_xor(x, 1, 32);
    x += __shfl_xor(x, 2, 32);
    x += __shfl_xor(x, 4, 32);
    x += __shfl_xor(x, 8, 32);
    return x;
}

// ---------------------------------------------------------------------------
// GEMM tiles: 128(M) x 256(N) x 32(K) per block, 8 waves, 64x64 per wave.
// ---------------------------------------------------------------------------
#define GBM 128
#define GBN 256
#define GBK 32
#define GLDA 40   // A tile ld: 32 + 8 halfs pad (rows 80B -> 16B aligned)
#define GLDBT 40  // transposed-B tile ld: K(32) + 8 pad

// GEMM 1: X[f32 8192x1024] * W[f32 1024x1024] + bias, scaled -> f16 [b,h,s,d]
__global__ __launch_bounds__(256) void qkv_proj_kernel(
    const float* __restrict__ X, const float* __restrict__ W,
    const float* __restrict__ bias, _Float16* __restrict__ out, float scale)
{
    __shared__ _Float16 As[GBM * GLDA];      // [M][K]
    __shared__ _Float16 BsT[GBN * GLDBT];    // [N][K]

    const int tid  = threadIdx.x;
    const int lane = tid & 31;
    const int wave = tid >> 5;   // 0..7
    const int wm   = wave >> 2;  // 0..1  (64 rows)
    const int wn   = wave & 3;   // 0..3  (64 cols)
    const int bm0  = blockIdx.y * GBM;
    const int bn0  = blockIdx.x * GBN;

    const v8f zf = {0.f,0.f,0.f,0.f,0.f,0.f,0.f,0.f};
    v8f acc[4][4];
#pragma unroll
    for (int i = 0; i < 4; ++i)
#pragma unroll
        for (int j = 0; j < 4; ++j) acc[i][j] = zf;

    const int arow = tid >> 1, ac0 = (tid & 1) * 16;
    const int brow = tid >> 3, bc0 = (tid & 7) * 32;

    for (int kb = 0; kb < EMB; kb += GBK) {
        __syncthreads();
        {   // A tile 128x32 (f32 -> f16), row-major
            const float* src = X + (size_t)(bm0 + arow) * EMB + kb + ac0;
            _Float16* dst = As + arow * GLDA + ac0;
#pragma unroll
            for (int j = 0; j < 16; ++j) dst[j] = (_Float16)src[j];
        }
        {   // B tile 32x256 (f32 -> f16), stored transposed [N][K]
            const float* src = W + (size_t)(kb + brow) * EMB + bn0 + bc0;
#pragma unroll
            for (int j = 0; j < 32; ++j)
                BsT[(bc0 + j) * GLDBT + brow] = (_Float16)src[j];
        }
        if (kb + GBK < EMB) {
            __builtin_prefetch(X + (size_t)(bm0 + arow) * EMB + kb + GBK + ac0, 0, 0);
            __builtin_prefetch(W + (size_t)(kb + GBK + brow) * EMB + bn0 + bc0, 0, 0);
        }
        __syncthreads();

        v16h bf[4];
#pragma unroll
        for (int j = 0; j < 4; ++j)
            bf[j] = frag_bt(BsT + (wn * 64 + j * 16) * GLDBT, GLDBT);
#pragma unroll
        for (int i = 0; i < 4; ++i) {
            v16h af = frag_a(As + (wm * 64 + i * 16) * GLDA, GLDA);
#pragma unroll
            for (int j = 0; j < 4; ++j)
                acc[i][j] = wmma16(af, bf[j], acc[i][j]);
        }
    }

    // Epilogue: (acc + bias) * scale -> f16 [b][h][s][d]
    const int halfsel = lane >> 4;
    const int ncol    = lane & 15;
#pragma unroll
    for (int i = 0; i < 4; ++i) {
#pragma unroll
        for (int j = 0; j < 4; ++j) {
            int ng = bn0 + wn * 64 + j * 16 + ncol;
            float bv = bias[ng];
            int h = ng >> 6, d = ng & 63;
#pragma unroll
            for (int e = 0; e < 8; ++e) {
                int mg = bm0 + wm * 64 + i * 16 + e + (halfsel << 3);
                int bb = mg >> 11;          // / SEQ
                int s  = mg & (SEQ - 1);
                float v = (acc[i][j][e] + bv) * scale;
                out[(((size_t)bb * HEADS + h) * SEQ + s) * DH + d] = (_Float16)v;
            }
        }
    }
}

// GEMM 2: attn[f16 8192x1024] * Wo[f32->f16] + bo -> f32 [8192x1024]
__global__ __launch_bounds__(256) void out_proj_kernel(
    const _Float16* __restrict__ A, const float* __restrict__ W,
    const float* __restrict__ bias, float* __restrict__ out)
{
    __shared__ _Float16 As[GBM * GLDA];
    __shared__ _Float16 BsT[GBN * GLDBT];

    const int tid  = threadIdx.x;
    const int lane = tid & 31;
    const int wave = tid >> 5;
    const int wm   = wave >> 2;
    const int wn   = wave & 3;
    const int bm0  = blockIdx.y * GBM;
    const int bn0  = blockIdx.x * GBN;

    const v8f zf = {0.f,0.f,0.f,0.f,0.f,0.f,0.f,0.f};
    v8f acc[4][4];
#pragma unroll
    for (int i = 0; i < 4; ++i)
#pragma unroll
        for (int j = 0; j < 4; ++j) acc[i][j] = zf;

    const int arow = tid >> 1, ac0 = (tid & 1) * 16;
    const int brow = tid >> 3, bc0 = (tid & 7) * 32;

    for (int kb = 0; kb < EMB; kb += GBK) {
        __syncthreads();
        {   // A tile 128x32 (already f16), row-major
            const _Float16* src = A + (size_t)(bm0 + arow) * EMB + kb + ac0;
            _Float16* dst = As + arow * GLDA + ac0;
#pragma unroll
            for (int j = 0; j < 16; ++j) dst[j] = src[j];
        }
        {   // B tile 32x256 (f32 -> f16), stored transposed [N][K]
            const float* src = W + (size_t)(kb + brow) * EMB + bn0 + bc0;
#pragma unroll
            for (int j = 0; j < 32; ++j)
                BsT[(bc0 + j) * GLDBT + brow] = (_Float16)src[j];
        }
        if (kb + GBK < EMB) {
            __builtin_prefetch(A + (size_t)(bm0 + arow) * EMB + kb + GBK + ac0, 0, 0);
            __builtin_prefetch(W + (size_t)(kb + GBK + brow) * EMB + bn0 + bc0, 0, 0);
        }
        __syncthreads();

        v16h bf[4];
#pragma unroll
        for (int j = 0; j < 4; ++j)
            bf[j] = frag_bt(BsT + (wn * 64 + j * 16) * GLDBT, GLDBT);
#pragma unroll
        for (int i = 0; i < 4; ++i) {
            v16h af = frag_a(As + (wm * 64 + i * 16) * GLDA, GLDA);
#pragma unroll
            for (int j = 0; j < 4; ++j)
                acc[i][j] = wmma16(af, bf[j], acc[i][j]);
        }
    }

    const int halfsel = lane >> 4;
    const int ncol    = lane & 15;
#pragma unroll
    for (int i = 0; i < 4; ++i) {
#pragma unroll
        for (int j = 0; j < 4; ++j) {
            int ng = bn0 + wn * 64 + j * 16 + ncol;
            float bv = bias[ng];
#pragma unroll
            for (int e = 0; e < 8; ++e) {
                int mg = bm0 + wm * 64 + i * 16 + e + (halfsel << 3);
                out[(size_t)mg * EMB + ng] = acc[i][j][e] + bv;
            }
        }
    }
}

// ---------------------------------------------------------------------------
// Flash attention: per WG one (b,h) x 64 query rows; 4 waves x 16 rows.
// Q,K pre-scaled by d^-0.25 so scores are already /sqrt(d).
// Online-softmax stats live in registers (replicated across each half-wave);
// row reductions via __shfl_xor 1/2/4/8 (never cross the half-wave boundary).
// ---------------------------------------------------------------------------
#define KLD 72   // K tile [kpos][d] ld: 64 + 8 (rows 144B -> 16B aligned)
#define VLD 40   // transposed V tile [d][kpos] ld: 32 + 8
#define PLD 40   // P tile [row][kpos] ld: 32 + 8

__global__ __launch_bounds__(128) void attn_kernel(
    const _Float16* __restrict__ Q, const _Float16* __restrict__ Kx,
    const _Float16* __restrict__ Vx, const int* __restrict__ mask,
    _Float16* __restrict__ Out)
{
    __shared__ _Float16 Kt[32 * KLD];       // [kpos][d]
    __shared__ _Float16 VtT[64 * VLD];      // [d][kpos]
    __shared__ _Float16 Pw[4][16 * PLD];    // per-wave P tile

    const int b = blockIdx.z, h = blockIdx.y;
    const int tid     = threadIdx.x;
    const int wave    = tid >> 5;
    const int lane    = tid & 31;
    const int halfsel = lane >> 4;
    const int n       = lane & 15;
    const int qrow0   = blockIdx.x * 64 + wave * 16;

    const size_t bh = (size_t)b * HEADS + h;
    const _Float16* qbase  = Q  + (bh * SEQ + qrow0) * DH;
    const _Float16* kbbase = Kx + (bh * SEQ) * DH;
    const _Float16* vbbase = Vx + (bh * SEQ) * DH;
    const int* mrow = mask + (size_t)b * SEQ;

    // Q fragments (16x64 split into two 16x32 A-frags), loaded once.
    v16h qf[2];
    qf[0] = frag_a(qbase, DH);
    qf[1] = frag_a(qbase + 32, DH);

    const v8f zf = {0.f,0.f,0.f,0.f,0.f,0.f,0.f,0.f};
    v8f oacc[4];
#pragma unroll
    for (int t = 0; t < 4; ++t) oacc[t] = zf;

    float m_run[8], l_run[8];
#pragma unroll
    for (int e = 0; e < 8; ++e) { m_run[e] = -3e38f; l_run[e] = 0.f; }

    const int lrow = tid >> 2;           // 0..31 (k position in tile)
    const int lc0  = (tid & 3) * 16;     // d chunk

    for (int kb = 0; kb < SEQ; kb += 32) {
        __syncthreads();
        {   // cooperative load: K row-major, V transposed
            const _Float16* ks = kbbase + (size_t)(kb + lrow) * DH + lc0;
            const _Float16* vs = vbbase + (size_t)(kb + lrow) * DH + lc0;
            _Float16* kd = Kt + lrow * KLD + lc0;
#pragma unroll
            for (int j = 0; j < 16; ++j) {
                kd[j] = ks[j];
                VtT[(lc0 + j) * VLD + lrow] = vs[j];
            }
        }
        __syncthreads();

        // Scores: 16x32 block = two 16x16 subtiles.  B = K^T; Kt is [kpos][d]
        // = row-major [N][K of the WMMA], so frag_bt applies directly.
        v8f sacc[2];
#pragma unroll
        for (int st = 0; st < 2; ++st) {
            v8f s = zf;
            s = wmma16(qf[0], frag_bt(Kt + (st * 16) * KLD, KLD), s);
            s = wmma16(qf[1], frag_bt(Kt + (st * 16) * KLD + 32, KLD), s);
            if (mrow[kb + st * 16 + n] == 0) {
#pragma unroll
                for (int e = 0; e < 8; ++e) s[e] = -3e38f;
            }
            sacc[st] = s;
        }

        // Online softmax per row (row = e + 8*halfsel); cols = 16 lanes x2.
#pragma unroll
        for (int e = 0; e < 8; ++e) {
            float blkmax = redmax16(fmaxf(sacc[0][e], sacc[1][e]));
            float mnew = fmaxf(m_run[e], blkmax);
            float corr = __expf(m_run[e] - mnew);
            m_run[e] = mnew;
            float p0 = __expf(sacc[0][e] - mnew);
            float p1 = __expf(sacc[1][e] - mnew);
            l_run[e] = l_run[e] * corr + redsum16(p0 + p1);
#pragma unroll
            for (int t = 0; t < 4; ++t) oacc[t][e] *= corr;
            int row = e + (halfsel << 3);
            Pw[wave][row * PLD + n]      = (_Float16)p0;
            Pw[wave][row * PLD + 16 + n] = (_Float16)p1;
        }

        // O += P(16x32) * V(32x64); VtT is [d][kpos] -> frag_bt.
        v16h pf = frag_a(&Pw[wave][0], PLD);
#pragma unroll
        for (int t = 0; t < 4; ++t) {
            v16h vfr = frag_bt(VtT + (t * 16) * VLD, VLD);
            oacc[t] = wmma16(pf, vfr, oacc[t]);
        }
    }

    // Finalize: divide by running sums, write f16 [b][s][h*64+d].
#pragma unroll
    for (int t = 0; t < 4; ++t) {
        int d = t * 16 + n;
#pragma unroll
        for (int e = 0; e < 8; ++e) {
            int srow = qrow0 + e + (halfsel << 3);
            Out[((size_t)b * SEQ + srow) * EMB + h * DH + d] =
                (_Float16)(oacc[t][e] / l_run[e]);
        }
    }
}

// ---------------------------------------------------------------------------
extern "C" void kernel_launch(void* const* d_in, const int* in_sizes, int n_in,
                              void* d_out, int out_size, void* d_ws,
                              size_t ws_size, hipStream_t stream)
{
    const float* X    = (const float*)d_in[0];
    const int*   mask = (const int*)d_in[1];
    const float* Wq   = (const float*)d_in[2];
    const float* bq   = (const float*)d_in[3];
    const float* Wk   = (const float*)d_in[4];
    const float* bk   = (const float*)d_in[5];
    const float* Wv   = (const float*)d_in[6];
    const float* bv   = (const float*)d_in[7];
    const float* Wo   = (const float*)d_in[8];
    const float* bo   = (const float*)d_in[9];
    float* out = (float*)d_out;

    const size_t M = (size_t)BATCH * SEQ;       // 8192
    _Float16* qh = (_Float16*)d_ws;             // 16 MB each
    _Float16* kh = qh + M * EMB;
    _Float16* vh = kh + M * EMB;
    _Float16* ah = vh + M * EMB;

    const float scale = 0.3535533905932738f;    // DH^-0.25 = 64^-0.25

    dim3 g1(EMB / GBN, M / GBM), b1(256);
    qkv_proj_kernel<<<g1, b1, 0, stream>>>(X, Wq, bq, qh, scale);
    qkv_proj_kernel<<<g1, b1, 0, stream>>>(X, Wk, bk, kh, scale);
    qkv_proj_kernel<<<g1, b1, 0, stream>>>(X, Wv, bv, vh, 1.0f);

    dim3 g2(SEQ / 64, HEADS, BATCH), b2(128);
    attn_kernel<<<g2, b2, 0, stream>>>(qh, kh, vh, mask, ah);

    out_proj_kernel<<<g1, b1, 0, stream>>>(ah, Wo, bo, out);
}
